// SentVae_7739531067828
// MI455X (gfx1250) — matrix operand order repaired and checked by reading, not compiled, not run on hardware
//
#include <hip/hip_runtime.h>
#include <hip/hip_bf16.h>
#include <stdint.h>

typedef __bf16 bf16_t;
typedef __attribute__((ext_vector_type(16))) __bf16 v16bf;
typedef __attribute__((ext_vector_type(8)))  float  v8f;

#define DEV __device__ __forceinline__

// ---------------- problem constants ----------------
constexpr int T_  = 25;
constexpr int E_  = 353,  EP  = 384;   // pad to mult of 32 (bf16 WMMA K)
constexpr int H_  = 191,  HP  = 192;
constexpr int Z_  = 13;
constexpr int V_  = 8185, VP  = 8192;
constexpr int G4  = 764,  G4P = 768;   // 4*H gates
constexpr int START_ = 19;
constexpr int ROWS = 16;               // batch rows per workgroup

constexpr int KT_E = EP / 32;          // 12 k-tiles over E
constexpr int KT_H = HP / 32;          // 6  k-tiles over H
constexpr int NT_G = G4P / 16;         // 48 n-tiles over gates
constexpr int NT_V = VP / 16;          // 512 n-tiles over vocab

// ---------------- swizzle index helpers (ISA 7.12.2 layouts) ----------------
DEV unsigned lane_id() { return threadIdx.x & 31u; }

// A fragment (16x32 bf16): lane = m + 16*hh, hh=(k>>3)&1, elem i=(k&7)|((k&16)>>1)
DEV int a_swz_idx(int m, int k) {
  return (k >> 5) * 512 + ((m + (((k >> 3) & 1) << 4)) << 4) + ((k & 7) | ((k & 16) >> 1));
}
// C/D fragment (16x16 f32): lane = n + 16*(m>>3), elem j = m&7
DEV int cd_swz_idx(int m, int n) {
  return (n >> 4) * 256 + (((n & 15) + ((m >> 3) << 4)) << 3) + (m & 7);
}

// A-row loader for row-major bf16 source (used for gathered embedding rows):
// elem i in [0,8): K=h*8+i ; i in [8,16): K=16+h*8+(i-8) -- two contiguous 16B chunks
DEV v16bf load_a_row(const bf16_t* rowbase /* row m, already offset by h*8 */) {
  v16bf a;
#pragma unroll
  for (int i = 0; i < 8; ++i) a[i] = rowbase[i];
#pragma unroll
  for (int i = 0; i < 8; ++i) a[i + 8] = rowbase[16 + i];
  return a;
}
DEV v16bf load_frag_swz(const bf16_t* tile) {          // 32B contiguous per lane
  return *(const v16bf*)(tile + lane_id() * 16);
}
DEV v8f wmma_bf16(v16bf a, v16bf b, v8f c) {
  return __builtin_amdgcn_wmma_f32_16x16x32_bf16(false, a, false, b, (short)0, c,
                                                 false, false);
}

DEV float sigm(float x) { return 1.f / (1.f + __expf(-x)); }
DEV unsigned fkey(float f) {  // monotonic float -> uint
  unsigned u = __float_as_uint(f);
  return (u & 0x80000000u) ? ~u : (u | 0x80000000u);
}

// ---------------- cast / pad kernels ----------------
// row-major padded bf16 (embedding table: A rows gathered contiguously)
__global__ void cast_pad_kernel(const float* __restrict__ src, bf16_t* __restrict__ dst,
                                int rows, int cols, int rowsPad, int colsPad) {
  long long idx = (long long)blockIdx.x * blockDim.x + threadIdx.x;
  long long total = (long long)rowsPad * colsPad;
  if (idx >= total) return;
  int r = (int)(idx / colsPad), c = (int)(idx % colsPad);
  float v = (r < rows && c < cols) ? src[(long long)r * cols + c] : 0.f;
  dst[idx] = (bf16_t)v;
}

// B-operand weights swizzled into fragment-native layout:
// dst[((nt*kTiles + kt)*512) + lane*16 + i] = src[k][n],
//   k = kt*32 + 16*(i>>3) + 8*(lane>>4) + (i&7) ; n = nt*16 + (lane&15)
__global__ void cast_swz_b_kernel(const float* __restrict__ src, bf16_t* __restrict__ dst,
                                  int rows, int cols, int kTiles, int nTiles) {
  long long idx = (long long)blockIdx.x * blockDim.x + threadIdx.x;
  long long total = (long long)nTiles * kTiles * 512;
  if (idx >= total) return;
  int e = (int)(idx & 511);
  long long tile = idx >> 9;
  int kt = (int)(tile % kTiles);
  int nt = (int)(tile / kTiles);
  int lane = e >> 4, i = e & 15;
  int k = kt * 32 + ((i & 8) << 1) + ((lane >> 4) << 3) + (i & 7);
  int n = nt * 16 + (lane & 15);
  float v = (k < rows && n < cols) ? src[(long long)k * cols + n] : 0.f;
  dst[idx] = (bf16_t)v;
}

__global__ void pad_bias_kernel(const float* __restrict__ src, float* __restrict__ dst,
                                int n, int npad, float padval) {
  int i = blockIdx.x * blockDim.x + threadIdx.x;
  if (i < npad) dst[i] = (i < n) ? src[i] : padval;
}

// ---------------- time-parallel input GEMM: xk = emb[inputs] @ inf_k + inf_b ----
// Output stored in C/D-fragment order grouped to match the rnn kernel's tiles:
// xk[((rg*T + t)*NT_G + nt)*256 + lane*8 + j]
__global__ void __launch_bounds__(32)
xk_kernel(const int* __restrict__ tokens, const bf16_t* __restrict__ emb_bf,
          const bf16_t* __restrict__ k_swz, const float* __restrict__ bias,
          float* __restrict__ xk_swz) {
  const int rg = blockIdx.x / T_;       // batch row-group (16 rows)
  const int t  = blockIdx.x % T_;
  const int nt = blockIdx.y;
  const unsigned l = lane_id();
  const int mA = l & 15, hA = l >> 4;
  const bf16_t* arow = emb_bf + (size_t)tokens[(rg * 16 + mA) * T_ + t] * EP + hA * 8;
  int n = nt * 16 + mA;
  float bv = (n < G4) ? bias[n] : 0.f;
  v8f acc;
#pragma unroll
  for (int j = 0; j < 8; ++j) acc[j] = bv;
  for (int kt = 0; kt < KT_E; ++kt) {
    v16bf a = load_a_row(arow + kt * 32);
    v16bf b = load_frag_swz(k_swz + ((size_t)nt * KT_E + kt) * 512);
    acc = wmma_bf16(a, b, acc);
  }
  *(v8f*)(xk_swz + (((size_t)rg * T_ + t) * NT_G + nt) * 256 + l * 8) = acc;
}

// ---------------- fused recurrent kernel ----------------
DEV void lstm_update(float* gates, float* c, bf16_t* h_swz, int tid) {
  for (int i = tid; i < ROWS * H_; i += 256) {
    int m = i / H_, k = i % H_;
    float gi = gates[cd_swz_idx(m, k)];
    float gf = gates[cd_swz_idx(m, H_ + k)];
    float gg = gates[cd_swz_idx(m, 2 * H_ + k)];
    float go = gates[cd_swz_idx(m, 3 * H_ + k)];
    float cn = sigm(gf) * c[m * HP + k] + sigm(gi) * tanhf(gg);
    c[m * HP + k] = cn;
    h_swz[a_swz_idx(m, k)] = (bf16_t)(sigm(go) * tanhf(cn));
  }
}

__global__ void __launch_bounds__(256)
rnn_kernel(const float* __restrict__ xk_swz, const bf16_t* __restrict__ inf_rk_swz,
           const float* __restrict__ eps,
           const float* __restrict__ mu_w, const float* __restrict__ mu_b,
           const float* __restrict__ sig_w, const float* __restrict__ sig_b,
           const float* __restrict__ init_w, const float* __restrict__ init_b,
           const bf16_t* __restrict__ emb_bf,
           const bf16_t* __restrict__ gen_k_swz, const bf16_t* __restrict__ gen_rk_swz,
           const float* __restrict__ gen_b,
           const bf16_t* __restrict__ out_w_swz, const float* __restrict__ out_b_pad,
           float* __restrict__ out) {
  __shared__ __align__(32) float  s_gates[NT_G * 256];   // 48 KB, C/D-fragment order
  __shared__ __align__(32) bf16_t s_h[KT_H * 512];       //  6 KB, A-fragment order
  __shared__ __align__(32) float  s_c[ROWS * HP];        // 12 KB
  __shared__ __align__(32) bf16_t s_x[KT_E * 512];       // 12 KB, A-fragment order
  __shared__ float s_z[ROWS][16];
  __shared__ unsigned long long s_best[ROWS];

  const int row0 = blockIdx.x * ROWS;
  const int tid  = threadIdx.x;
  const int wave = tid >> 5;
  const unsigned l = tid & 31;
  const int mA = l & 15, hA = l >> 4;
  const float* xkbase = xk_swz + (size_t)blockIdx.x * T_ * NT_G * 256;

  for (int i = tid; i < KT_H * 512; i += 256) s_h[i] = (bf16_t)0.f;
  for (int i = tid; i < ROWS * HP; i += 256) s_c[i] = 0.f;
  __syncthreads();

  // ================= encoder: 25 sequential steps =================
  for (int t = 0; t < T_; ++t) {
    for (int it = 0; it < 6; ++it) {                      // 48 n-tiles / 8 waves
      int nt = wave * 6 + it;
      v8f acc = *(const v8f*)(xkbase + ((size_t)t * NT_G + nt) * 256 + l * 8);
#pragma unroll
      for (int kt = 0; kt < KT_H; ++kt) {                 // + h @ rk
        v16bf a = load_frag_swz(s_h + kt * 512);
        v16bf b = load_frag_swz(inf_rk_swz + ((size_t)nt * KT_H + kt) * 512);
        acc = wmma_bf16(a, b, acc);
      }
      *(v8f*)(s_gates + nt * 256 + l * 8) = acc;
    }
    __syncthreads();
    lstm_update(s_gates, s_c, s_h, tid);
    __syncthreads();
  }

  // ================= z = mu + eps*sigma ; init decoder state =================
  for (int i = tid; i < ROWS * Z_; i += 256) {
    int m = i / Z_, zi = i % Z_;
    float mu = mu_b[zi], sg = sig_b[zi];
    for (int k = 0; k < H_; ++k) {
      float hv = (float)s_h[a_swz_idx(m, k)];
      mu += hv * mu_w[k * Z_ + zi];
      sg += hv * sig_w[k * Z_ + zi];
    }
    s_z[m][zi] = mu + eps[zi] * sg;
  }
  __syncthreads();
  for (int i = tid; i < ROWS * HP; i += 256) {
    int m = i / HP, k = i % HP;
    float v = 0.f;
    if (k < H_) {
      v = init_b[k];
      for (int zi = 0; zi < Z_; ++zi) v += s_z[m][zi] * init_w[zi * H_ + k];
    }
    s_c[m * HP + k] = v;                                  // c0 = h0 = init_state
    s_h[a_swz_idx(m, k)] = (bf16_t)v;
  }
  for (int i = tid; i < ROWS * EP; i += 256) {            // x0 = emb[START]
    int m = i / EP, e = i % EP;
    s_x[a_swz_idx(m, e)] = emb_bf[(size_t)START_ * EP + e];
  }
  __syncthreads();

  // ================= autoregressive decoder: 25 sequential steps =================
  for (int t = 0; t < T_; ++t) {
    if (tid < ROWS) s_best[tid] = 0ull;
    // gates = x@gen_k + h@gen_rk + gen_b
    for (int it = 0; it < 6; ++it) {
      int nt = wave * 6 + it;
      float bv = (nt * 16 + mA < G4) ? gen_b[nt * 16 + mA] : 0.f;
      v8f acc;
#pragma unroll
      for (int j = 0; j < 8; ++j) acc[j] = bv;
#pragma unroll
      for (int kt = 0; kt < KT_E; ++kt) {
        v16bf a = load_frag_swz(s_x + kt * 512);
        v16bf b = load_frag_swz(gen_k_swz + ((size_t)nt * KT_E + kt) * 512);
        acc = wmma_bf16(a, b, acc);
      }
#pragma unroll
      for (int kt = 0; kt < KT_H; ++kt) {
        v16bf a = load_frag_swz(s_h + kt * 512);
        v16bf b = load_frag_swz(gen_rk_swz + ((size_t)nt * KT_H + kt) * 512);
        acc = wmma_bf16(a, b, acc);
      }
      *(v8f*)(s_gates + nt * 256 + l * 8) = acc;
    }
    __syncthreads();
    lstm_update(s_gates, s_c, s_h, tid);
    __syncthreads();

    // logits = h @ out_w + out_b ; per-lane register argmax, 8 atomics at the end
    unsigned bk[8];
    int bn[8];
#pragma unroll
    for (int j = 0; j < 8; ++j) { bk[j] = 0u; bn[j] = 0; }
    for (int tile = wave; tile < NT_V; tile += 8) {
      int n0 = tile * 16;
      float bv = out_b_pad[n0 + mA];
      v8f acc;
#pragma unroll
      for (int j = 0; j < 8; ++j) acc[j] = bv;
#pragma unroll
      for (int kt = 0; kt < KT_H; ++kt) {
        v16bf a = load_frag_swz(s_h + kt * 512);
        v16bf b = load_frag_swz(out_w_swz + ((size_t)tile * KT_H + kt) * 512);
        acc = wmma_bf16(a, b, acc);
      }
      if (tile + 8 < NT_V)                                 // hide L2 latency
        __builtin_prefetch(out_w_swz + ((size_t)(tile + 8) * KT_H) * 512 + l * 16, 0, 0);
      int n = n0 + mA;
      if (n < V_) {
#pragma unroll
        for (int j = 0; j < 8; ++j) {
          int m = j + 8 * hA;
          float v = acc[j];
          out[((size_t)(row0 + m) * T_ + t) * V_ + n] = v;
          unsigned kk = fkey(v);
          if (kk > bk[j]) { bk[j] = kk; bn[j] = n; }       // strict > : first max wins
        }
      }
    }
#pragma unroll
    for (int j = 0; j < 8; ++j) {                          // ties -> smallest index
      unsigned long long key =
          ((unsigned long long)bk[j] << 32) | (unsigned)(0xFFFFFFFFu - (unsigned)bn[j]);
      atomicMax(&s_best[j + 8 * hA], key);
    }
    __syncthreads();
    // x_next = emb[argmax]
    for (int i = tid; i < ROWS * EP; i += 256) {
      int m = i / EP, e = i % EP;
      unsigned idx = 0xFFFFFFFFu - (unsigned)(s_best[m] & 0xFFFFFFFFu);
      s_x[a_swz_idx(m, e)] = emb_bf[(size_t)idx * EP + e];
    }
    __syncthreads();
  }
}

// ---------------- launcher ----------------
extern "C" void kernel_launch(void* const* d_in, const int* in_sizes, int n_in,
                              void* d_out, int out_size, void* d_ws, size_t ws_size,
                              hipStream_t stream) {
  const int*   inputs = (const int*)d_in[0];
  const float* eps    = (const float*)d_in[1];
  const float* emb    = (const float*)d_in[2];
  const float* inf_k  = (const float*)d_in[3];
  const float* inf_rk = (const float*)d_in[4];
  const float* inf_b  = (const float*)d_in[5];
  const float* mu_w   = (const float*)d_in[6];
  const float* mu_b   = (const float*)d_in[7];
  const float* sig_w  = (const float*)d_in[8];
  const float* sig_b  = (const float*)d_in[9];
  const float* init_w = (const float*)d_in[10];
  const float* init_b = (const float*)d_in[11];
  const float* gen_k  = (const float*)d_in[12];
  const float* gen_rk = (const float*)d_in[13];
  const float* gen_b  = (const float*)d_in[14];
  const float* out_w  = (const float*)d_in[15];
  const float* out_b  = (const float*)d_in[16];
  float* out = (float*)d_out;

  uintptr_t p = (uintptr_t)d_ws;
  auto carve = [&p](size_t bytes) -> void* {
    void* r = (void*)p;
    p += (bytes + 255) & ~(size_t)255;
    return r;
  };
  bf16_t* emb_bf     = (bf16_t*)carve((size_t)V_ * EP * 2);
  bf16_t* inf_k_swz  = (bf16_t*)carve((size_t)NT_G * KT_E * 512 * 2);
  bf16_t* inf_rk_swz = (bf16_t*)carve((size_t)NT_G * KT_H * 512 * 2);
  bf16_t* gen_k_swz  = (bf16_t*)carve((size_t)NT_G * KT_E * 512 * 2);
  bf16_t* gen_rk_swz = (bf16_t*)carve((size_t)NT_G * KT_H * 512 * 2);
  bf16_t* out_w_swz  = (bf16_t*)carve((size_t)NT_V * KT_H * 512 * 2);
  float*  out_b_pad  = (float*)carve((size_t)VP * 4);
  float*  xk_swz     = (float*)carve((size_t)16 * T_ * NT_G * 256 * 4);

  auto cgrid = [](long long n) { return dim3((unsigned)((n + 255) / 256)); };
  cast_pad_kernel<<<cgrid((long long)V_ * EP), 256, 0, stream>>>(emb, emb_bf, V_, E_, V_, EP);
  cast_swz_b_kernel<<<cgrid((long long)NT_G * KT_E * 512), 256, 0, stream>>>(
      inf_k, inf_k_swz, E_, G4, KT_E, NT_G);
  cast_swz_b_kernel<<<cgrid((long long)NT_G * KT_H * 512), 256, 0, stream>>>(
      inf_rk, inf_rk_swz, H_, G4, KT_H, NT_G);
  cast_swz_b_kernel<<<cgrid((long long)NT_G * KT_E * 512), 256, 0, stream>>>(
      gen_k, gen_k_swz, E_, G4, KT_E, NT_G);
  cast_swz_b_kernel<<<cgrid((long long)NT_G * KT_H * 512), 256, 0, stream>>>(
      gen_rk, gen_rk_swz, H_, G4, KT_H, NT_G);
  cast_swz_b_kernel<<<cgrid((long long)NT_V * KT_H * 512), 256, 0, stream>>>(
      out_w, out_w_swz, H_, V_, KT_H, NT_V);
  pad_bias_kernel<<<cgrid(VP), 256, 0, stream>>>(out_b, out_b_pad, V_, VP, -1e30f);

  dim3 xg(16 * T_, NT_G);
  xk_kernel<<<xg, 32, 0, stream>>>(inputs, emb_bf, inf_k_swz, inf_b, xk_swz);

  rnn_kernel<<<16, 256, 0, stream>>>(xk_swz, inf_rk_swz, eps, mu_w, mu_b, sig_w, sig_b,
                                     init_w, init_b, emb_bf, gen_k_swz, gen_rk_swz,
                                     gen_b, out_w_swz, out_b_pad, out);
}
